// HEALEncoder_40518721470589
// MI455X (gfx1250) — matrix operand order repaired and checked by reading, not compiled, not run on hardware
//
#include <hip/hip_runtime.h>
#include <hip/hip_bf16.h>
#include <math.h>

// ---------------- types ----------------
typedef __bf16 bf16;
typedef __attribute__((ext_vector_type(8)))  bf16  v8bf;
typedef __attribute__((ext_vector_type(16))) bf16  v16bf;
typedef __attribute__((ext_vector_type(8)))  float v8f;

// ---------------- problem constants ----------------
#define BATCH   2
#define NEDGE   393216
#define NREC    12288
#define DX      128
#define DEDGE   4
#define K1      160      // 132 padded to mult of 32
#define K1REAL  132
#define NOUT    256
#define KP1     168      // LDS row stride (bf16) for K=160 region (84 dwords, 84%64=20 -> conflict free)
#define KP2     264      // LDS row stride for K=256 region (132 dwords, 132%64=4 -> conflict free)
#define MTILE   64       // rows per workgroup tile
#define NTILES1 ((BATCH*NEDGE)/MTILE)   // 12288
#define NTILES3 ((BATCH*NREC)/MTILE)    // 384
#define GRID1   2048

// ---------------- LDS partition (dynamic shared) ----------------
#define S1_WT1_OFF  0
#define S1_WT2_OFF  (256*KP1*2)                       // 86016
#define S1_AH_OFF   (S1_WT2_OFF + 256*KP2*2)          // 221184
#define S1_RECV_OFF (S1_AH_OFF + MTILE*KP2*2)         // 254976
#define S1_SMEM     (S1_RECV_OFF + MTILE*4)           // 255232 bytes
#define S3_WT1_OFF  0
#define S3_WT2_OFF  (256*KP2*2)                       // 135168
#define S3_AH_OFF   (S3_WT2_OFF + 256*KP2*2)          // 270336
#define S3_SMEM     (S3_AH_OFF + MTILE*KP2*2)         // 304128 bytes

// ---------------- helpers ----------------
__device__ __forceinline__ float siluf(float v) {
    return v / (1.0f + __expf(-v));
}

__device__ __forceinline__ void atomAddF(float* p, float v) {
    __hip_atomic_fetch_add(p, v, __ATOMIC_RELAXED, __HIP_MEMORY_SCOPE_AGENT);
}

// A-fragment (16x32 bf16): lanes 0-15 row m, K {0..7,16..23}; lanes 16-31 row m, K {8..15,24..31}
__device__ __forceinline__ v16bf ldsAfrag(const bf16* mat, int row0, int kbase, int stride) {
    const int lane = threadIdx.x & 31;
    const bf16* p = mat + (row0 + (lane & 15)) * stride + kbase + ((lane >> 4) << 3);
    v8bf lo = *(const v8bf*)(p);
    v8bf hi = *(const v8bf*)(p + 16);
    return __builtin_shufflevector(lo, hi, 0,1,2,3,4,5,6,7,8,9,10,11,12,13,14,15);
}

// B-fragment (32x16 bf16), weights stored transposed [N][K]:
// lanes 0-15 col n, K {0..7,16..23}; lanes 16-31 col n, K {8..15,24..31}
__device__ __forceinline__ v16bf ldsBfrag(const bf16* mat, int ncol0, int kbase, int stride) {
    const int lane = threadIdx.x & 31;
    const bf16* p = mat + (ncol0 + (lane & 15)) * stride + kbase + ((lane >> 4) << 3);
    v8bf lo = *(const v8bf*)(p);
    v8bf hi = *(const v8bf*)(p + 16);
    return __builtin_shufflevector(lo, hi, 0,1,2,3,4,5,6,7,8,9,10,11,12,13,14,15);
}

__device__ __forceinline__ v8f wmma_bf16(v16bf a, v16bf b, v8f c) {
    return __builtin_amdgcn_wmma_f32_16x16x32_bf16(false, a, false, b, (short)0, c, false, false);
}

// One k-step of a 16x128 output block: batch-issue all LDS fragment loads before
// the 8 WMMAs so the ds waits overlap matrix ops instead of serializing.
__device__ __forceinline__ void gemm_kstep(const bf16* Amat, int wm, int kk, int astr,
                                           const bf16* Bmat, int wn, int bstr, v8f acc[8]) {
    v16bf af = ldsAfrag(Amat, wm, kk, astr);
    v16bf bfr[8];
    #pragma unroll
    for (int nt = 0; nt < 8; nt++) bfr[nt] = ldsBfrag(Bmat, wn + nt * 16, kk, bstr);
    #pragma unroll
    for (int nt = 0; nt < 8; nt++) acc[nt] = wmma_bf16(af, bfr[nt], acc[nt]);
}

// ---------------- kernel: zero node accumulator ----------------
__global__ void zero_kernel(float4* p, int n4) {
    int stride = blockDim.x * gridDim.x;
    float4 z = make_float4(0.f, 0.f, 0.f, 0.f);
    for (int i = blockIdx.x * blockDim.x + threadIdx.x; i < n4; i += stride) p[i] = z;
}

// ---------------- kernel: convert weights fp32 -> bf16, transposed [N][K] ----------------
__global__ void prep_weights(const float* __restrict__ w1e, const float* __restrict__ w2e,
                             const float* __restrict__ w1f, const float* __restrict__ w2f,
                             bf16* __restrict__ Wt1, bf16* __restrict__ Wt2,
                             bf16* __restrict__ Wt1f, bf16* __restrict__ Wt2f) {
    int stride = blockDim.x * gridDim.x;
    int t0 = blockIdx.x * blockDim.x + threadIdx.x;
    for (int f = t0; f < 256 * K1; f += stride) {       // Wt1[n][k], K padded 132->160 with zeros
        int n = f / K1, k = f % K1;
        Wt1[f] = (k < K1REAL) ? (bf16)w1e[k * 256 + n] : (bf16)0.0f;
    }
    for (int f = t0; f < 256 * 256; f += stride) {
        int n = f >> 8, k = f & 255;
        Wt2[f]  = (bf16)w2e[k * 256 + n];
        Wt1f[f] = (bf16)w1f[k * 256 + n];
        Wt2f[f] = (bf16)w2f[k * 256 + n];
    }
}

// ---------------- kernel: fused edge MLP + scatter-add ----------------
__global__ __launch_bounds__(256) void edge_mlp_scatter(
    const float* __restrict__ x, const float* __restrict__ edge_attr,
    const int* __restrict__ edge_index,
    const float* __restrict__ b1e, const float* __restrict__ b2e,
    const bf16* __restrict__ gWt1, const bf16* __restrict__ gWt2,
    float* __restrict__ vmsum) {
    extern __shared__ char smem[];
    bf16* lWt1 = (bf16*)(smem + S1_WT1_OFF);
    bf16* lWt2 = (bf16*)(smem + S1_WT2_OFF);
    bf16* A1   = (bf16*)(smem + S1_AH_OFF);   // [64][KP1]
    bf16* H    = (bf16*)(smem + S1_AH_OFF);   // [64][KP2] (union with A1)
    int*  recv = (int*)(smem + S1_RECV_OFF);

    const int tid  = threadIdx.x;
    const int lane = tid & 31;
    const int w    = tid >> 5;
    const int wm   = (w & 3) * 16;     // wave's M-row base within tile
    const int wn   = (w >> 2) * 128;   // wave's N-col base

    // cache weights in LDS once per block (L2-resident source)
    for (int f = tid; f < 256 * (K1 / 8); f += 256) {   // Wt1: 20 chunks/row
        int r = f / 20, c8 = f % 20;
        *(v8bf*)(lWt1 + r * KP1 + c8 * 8) = *(const v8bf*)(gWt1 + r * K1 + c8 * 8);
    }
    for (int f = tid; f < 256 * 32; f += 256) {         // Wt2: 32 chunks/row
        int r = f >> 5, c8 = f & 31;
        *(v8bf*)(lWt2 + r * KP2 + c8 * 8) = *(const v8bf*)(gWt2 + r * 256 + c8 * 8);
    }
    // per-lane bias registers for the 8 N-subtiles this wave owns
    float bias1[8], bias2[8];
    #pragma unroll
    for (int nt = 0; nt < 8; nt++) {
        int n = wn + nt * 16 + (lane & 15);
        bias1[nt] = b1e[n];
        bias2[nt] = b2e[n];
    }
    __syncthreads();

    for (int tile = blockIdx.x; tile < NTILES1; tile += gridDim.x) {
        const long long rowBase = (long long)tile * MTILE;
        const int b  = (int)(rowBase / NEDGE);
        const int e0 = (int)(rowBase % NEDGE);   // tiles never straddle batches (NEDGE%64==0)

        // ---- build A = [edge_attr | x | 0] as bf16 ----
        if (tid < MTILE) recv[tid] = edge_index[NEDGE + e0 + tid];
        {   int r = tid >> 2, c = tid & 3;       // 64 rows x 4 edge_attr cols
            A1[r * KP1 + c] = (bf16)edge_attr[(size_t)(e0 + r) * DEDGE + c];
        }
        const float* xb = x + ((size_t)b * NEDGE + e0) * DX;
        #pragma unroll
        for (int j = 0; j < 8; j++) {            // 64 rows x 32 float4 of x
            int f = j * 256 + tid;
            int r = f >> 5, c4 = f & 31;
            float4 v = ((const float4*)(xb + (size_t)r * DX))[c4];
            bf16* d = A1 + r * KP1 + DEDGE + c4 * 4;
            d[0] = (bf16)v.x; d[1] = (bf16)v.y; d[2] = (bf16)v.z; d[3] = (bf16)v.w;
        }
        for (int f = tid; f < MTILE * (K1 - K1REAL); f += 256) {   // zero K pad
            int r = f / (K1 - K1REAL), c = K1REAL + f % (K1 - K1REAL);
            A1[r * KP1 + c] = (bf16)0.0f;
        }
        __syncthreads();

        // prefetch next tile's x slab (64 rows x 512B = 32KB; 256 threads x 128B)
        {
            int nextTile = tile + gridDim.x;
            if (nextTile < NTILES1) {
                long long rb2 = (long long)nextTile * MTILE;
                int b2 = (int)(rb2 / NEDGE), e02 = (int)(rb2 % NEDGE);
                const char* nx = (const char*)(x + ((size_t)b2 * NEDGE + e02) * DX);
                __builtin_prefetch(nx + (size_t)tid * 128, 0, 0);
            }
        }

        // ---- layer 1: [64x160] x [160x256] ----
        v8f acc[8];
        #pragma unroll
        for (int nt = 0; nt < 8; nt++) acc[nt] = (v8f){};
        #pragma unroll
        for (int kk = 0; kk < K1; kk += 32)
            gemm_kstep(A1, wm, kk, KP1, lWt1, wn, KP1, acc);
        __syncthreads();   // all waves done reading A1 before H overwrites it

        // ---- bias + SiLU -> H (bf16) ----
        #pragma unroll
        for (int nt = 0; nt < 8; nt++) {
            #pragma unroll
            for (int r = 0; r < 8; r++) {
                int m   = ((lane >> 4) << 3) + r;
                int n   = wn + nt * 16 + (lane & 15);
                float v = acc[nt][r] + bias1[nt];
                H[(wm + m) * KP2 + n] = (bf16)siluf(v);
            }
        }
        __syncthreads();

        // ---- layer 2: [64x256] x [256x256] ----
        #pragma unroll
        for (int nt = 0; nt < 8; nt++) acc[nt] = (v8f){};
        #pragma unroll
        for (int kk = 0; kk < 256; kk += 32)
            gemm_kstep(H, wm, kk, KP2, lWt2, wn, KP2, acc);

        // ---- scatter-add into receiver nodes (L2-resident accumulator) ----
        float* vmb = vmsum + (size_t)b * NREC * NOUT;
        #pragma unroll
        for (int nt = 0; nt < 8; nt++) {
            #pragma unroll
            for (int r = 0; r < 8; r++) {
                int m    = ((lane >> 4) << 3) + r;
                int n    = wn + nt * 16 + (lane & 15);
                int node = recv[wm + m];
                atomAddF(&vmb[(size_t)node * NOUT + n], acc[nt][r] + bias2[nt]);
            }
        }
        __syncthreads();   // protect A1/H/recv before next tile
    }
}

// ---------------- kernel: node MLP ----------------
__global__ __launch_bounds__(256) void node_mlp(
    const float* __restrict__ vmsum,
    const float* __restrict__ b1f, const float* __restrict__ b2f,
    const bf16* __restrict__ gWt1, const bf16* __restrict__ gWt2,
    float* __restrict__ out) {
    extern __shared__ char smem[];
    bf16* lWt1 = (bf16*)(smem + S3_WT1_OFF);
    bf16* lWt2 = (bf16*)(smem + S3_WT2_OFF);
    bf16* A1   = (bf16*)(smem + S3_AH_OFF);   // [64][KP2]
    bf16* H    = (bf16*)(smem + S3_AH_OFF);   // union

    const int tid  = threadIdx.x;
    const int lane = tid & 31;
    const int w    = tid >> 5;
    const int wm   = (w & 3) * 16;
    const int wn   = (w >> 2) * 128;

    for (int f = tid; f < 256 * 32; f += 256) {
        int r = f >> 5, c8 = f & 31;
        *(v8bf*)(lWt1 + r * KP2 + c8 * 8) = *(const v8bf*)(gWt1 + r * 256 + c8 * 8);
        *(v8bf*)(lWt2 + r * KP2 + c8 * 8) = *(const v8bf*)(gWt2 + r * 256 + c8 * 8);
    }
    float bias1[8], bias2[8];
    #pragma unroll
    for (int nt = 0; nt < 8; nt++) {
        int n = wn + nt * 16 + (lane & 15);
        bias1[nt] = b1f[n];
        bias2[nt] = b2f[n];
    }
    __syncthreads();

    for (int tile = blockIdx.x; tile < NTILES3; tile += gridDim.x) {
        const size_t rowBase = (size_t)tile * MTILE;

        const float* src = vmsum + rowBase * NOUT;
        #pragma unroll
        for (int j = 0; j < 16; j++) {           // 64 rows x 64 float4
            int f = j * 256 + tid;
            int r = f >> 6, c4 = f & 63;
            float4 v = ((const float4*)(src + (size_t)r * NOUT))[c4];
            bf16* d = A1 + r * KP2 + c4 * 4;
            d[0] = (bf16)v.x; d[1] = (bf16)v.y; d[2] = (bf16)v.z; d[3] = (bf16)v.w;
        }
        __syncthreads();

        v8f acc[8];
        #pragma unroll
        for (int nt = 0; nt < 8; nt++) acc[nt] = (v8f){};
        #pragma unroll
        for (int kk = 0; kk < 256; kk += 32)
            gemm_kstep(A1, wm, kk, KP2, lWt1, wn, KP2, acc);
        __syncthreads();

        #pragma unroll
        for (int nt = 0; nt < 8; nt++) {
            #pragma unroll
            for (int r = 0; r < 8; r++) {
                int m   = ((lane >> 4) << 3) + r;
                int n   = wn + nt * 16 + (lane & 15);
                float v = acc[nt][r] + bias1[nt];
                H[(wm + m) * KP2 + n] = (bf16)siluf(v);
            }
        }
        __syncthreads();

        #pragma unroll
        for (int nt = 0; nt < 8; nt++) acc[nt] = (v8f){};
        #pragma unroll
        for (int kk = 0; kk < 256; kk += 32)
            gemm_kstep(H, wm, kk, KP2, lWt2, wn, KP2, acc);

        #pragma unroll
        for (int nt = 0; nt < 8; nt++) {
            #pragma unroll
            for (int r = 0; r < 8; r++) {
                int m = ((lane >> 4) << 3) + r;
                int n = wn + nt * 16 + (lane & 15);
                out[(rowBase + wm + m) * NOUT + n] = acc[nt][r] + bias2[nt];
            }
        }
        __syncthreads();
    }
}

// ---------------- host launch ----------------
extern "C" void kernel_launch(void* const* d_in, const int* in_sizes, int n_in,
                              void* d_out, int out_size, void* d_ws, size_t ws_size,
                              hipStream_t stream) {
    (void)in_sizes; (void)n_in; (void)out_size; (void)ws_size;
    const float* x         = (const float*)d_in[0];
    const float* edge_attr = (const float*)d_in[1];
    const int*   edge_idx  = (const int*)d_in[2];
    const float* w1e = (const float*)d_in[3];
    const float* b1e = (const float*)d_in[4];
    const float* w2e = (const float*)d_in[5];
    const float* b2e = (const float*)d_in[6];
    const float* w1f = (const float*)d_in[7];
    const float* b1f = (const float*)d_in[8];
    const float* w2f = (const float*)d_in[9];
    const float* b2f = (const float*)d_in[10];
    float* out = (float*)d_out;

    // workspace layout
    char*  ws    = (char*)d_ws;
    float* vmsum = (float*)ws;                                   // [B][NREC][256] f32
    size_t off   = (size_t)BATCH * NREC * NOUT * sizeof(float);  // 25,165,824
    bf16* Wt1  = (bf16*)(ws + off); off += (size_t)256 * K1  * 2;
    bf16* Wt2  = (bf16*)(ws + off); off += (size_t)256 * 256 * 2;
    bf16* Wt1f = (bf16*)(ws + off); off += (size_t)256 * 256 * 2;
    bf16* Wt2f = (bf16*)(ws + off);

    hipFuncSetAttribute((const void*)edge_mlp_scatter,
                        hipFuncAttributeMaxDynamicSharedMemorySize, S1_SMEM);
    hipFuncSetAttribute((const void*)node_mlp,
                        hipFuncAttributeMaxDynamicSharedMemorySize, S3_SMEM);

    const int n4 = BATCH * NREC * NOUT / 4;
    zero_kernel<<<2048, 256, 0, stream>>>((float4*)vmsum, n4);
    prep_weights<<<256, 256, 0, stream>>>(w1e, w2e, w1f, w2f, Wt1, Wt2, Wt1f, Wt2f);
    edge_mlp_scatter<<<GRID1, 256, S1_SMEM, stream>>>(x, edge_attr, edge_idx,
                                                      b1e, b2e, Wt1, Wt2, vmsum);
    node_mlp<<<NTILES3, 256, S3_SMEM, stream>>>(vmsum, b1f, b2f, Wt1f, Wt2f, out);
}
